// NLSPN_76493367542203
// MI455X (gfx1250) — matrix operand order repaired and checked by reading, not compiled, not run on hardware
//
#include <hip/hip_runtime.h>
#include <hip/hip_bf16.h>
#include <math.h>

// ---------------- problem constants ----------------
constexpr int HH = 352;
constexpr int WW = 1216;
constexpr int HW = HH * WW;          // 428032 (divisible by 256 and 16)
constexpr int TSTEPS = 6;

typedef __attribute__((ext_vector_type(16))) _Float16 v16h;
typedef __attribute__((ext_vector_type(8)))  float    v8f;

union U16h { v16h v; _Float16 e[16]; };
union U8f  { v8f  v; float    e[8];  };

__device__ __forceinline__ int iclamp(int v, int lo, int hi) {
  return (v < lo) ? lo : ((v > hi) ? hi : v);
}

// =====================================================================
// Pack conv weights (aff 9ch + off 16ch, K=72 im2col) into a
// WMMA A-fragment lane layout (contiguous k order, same as B) for v_wmma_f32_16x16x32_f16.
// Layout idx = (((t*2+mtile)*3+kchunk)*32+lane)*16+elem
// A 16x32 f16: lane<16: elems 0..15 -> K 0..15
//              lane>=16: elems 0..15 -> K 16..31
// =====================================================================
__global__ __launch_bounds__(256) void nlspn_pack_weights(
    const float* __restrict__ aff_w_w,   // (6,9,8,3,3)
    const float* __restrict__ aff_o_w,   // (6,16,8,3,3)
    _Float16* __restrict__ wpk)
{
  int id = blockIdx.x * 256 + threadIdx.x;
  if (id >= TSTEPS * 2 * 3 * 32 * 16) return;
  int i    = id & 15;
  int frag = id >> 4;
  int lane = frag & 31;
  int rest = frag >> 5;        // (t*2+mt)*3 + kc
  int kc   = rest % 3;
  int tm   = rest / 3;
  int mt   = tm & 1;
  int t    = tm >> 1;

  int M = mt * 16 + (lane & 15);
  int K = kc * 32 + i + ((lane >= 16) ? 16 : 0);

  float v = 0.f;
  if (K < 72) {
    int c = K / 9, pos = K % 9;
    if (M < 9)       v = aff_w_w[((t * 9  +  M     ) * 8 + c) * 9 + pos];
    else if (M < 25) v = aff_o_w[((t * 16 + (M - 9)) * 8 + c) * 9 + pos];
  }
  wpk[id] = (_Float16)v;
}

// =====================================================================
// Branch-free B-fragment builder. KOFF = 0 (lanes 0..15) or 16.
// All k/c/pos/dy/dx are compile-time -> unconditional ds_load with
// immediate offsets; OOB handled by v_cndmask on the loaded value.
// LDS layout: sw[c*60 + r*20 + cl], cl <-> gx = x0 + cl - 1 (clamped).
// =====================================================================
template <int KOFF>
__device__ __forceinline__ void build_B(U16h (&bfr)[3], const float* sw,
                                        int n, int ty, int x)
{
#pragma unroll
  for (int kc = 0; kc < 3; ++kc) {
#pragma unroll
    for (int i = 0; i < 16; ++i) {
      const int k = kc * 32 + KOFF + i;       // compile-time
      float v = 0.f;
      if (k < 72) {                           // compile-time prune
        const int c = k / 9, pos = k % 9;
        const int dy = pos / 3 - 1, dx = pos % 3 - 1;
        float lv = sw[c * 60 + (pos / 3) * 20 + (pos % 3) + n];  // always safe
        int yy = ty + dy, xx = x + dx;
        bool ok = (yy >= 0) & (yy < HH) & (xx >= 0) & (xx < WW); // folds when dy==0/dx==0
        v = ok ? lv : 0.f;
      }
      bfr[kc].e[i] = (_Float16)v;
    }
  }
}

// =====================================================================
// One propagation step, fully fused:
//   async-copy guidance tile (8ch x 3rows x 20cols) to LDS   [ASYNCcnt]
//   -> WMMA conv3x3 (8ch -> 9 aff + 16 off)  ->  sigmoid/normalize
//   -> 9-tap deformable bilinear sample of feat_in -> feat_out
// One wave = one 16-pixel tile (row-aligned; WW % 16 == 0).
// =====================================================================
__global__ __launch_bounds__(256) void nlspn_step_kernel(
    const float* __restrict__ guidance,
    const float* __restrict__ feat_in,
    const _Float16* __restrict__ wpk,
    const float* __restrict__ aff_w_b,   // (6,9)
    const float* __restrict__ aff_o_b,   // (6,16)
    float* __restrict__ feat_out,
    int t)
{
  __shared__ float smem[8 * 480];             // 8 waves x (8ch*3row*20col)

  const int lane = threadIdx.x & 31;
  const int wid  = threadIdx.x >> 5;
  const int wave = (blockIdx.x * blockDim.x + threadIdx.x) >> 5;
  const int p0   = wave * 16;
  if (p0 >= HW) return;                       // wave-uniform
  const int n  = lane & 15;
  const int ty = p0 / WW;                     // tile row (uniform)
  const int x0 = p0 % WW;                     // tile start col (uniform)
  const int x  = x0 + n;
  const int p  = p0 + n;

  const float* g = guidance + (size_t)t * 8 * HW;
  const unsigned long long gbase = (unsigned long long)g;

  float* sw = &smem[wid * 480];
  const unsigned swbase = (unsigned)(unsigned long long)(void*)sw;

  // ---- stage guidance tile into LDS via async copy engine (ASYNCcnt) ----
  // layout q = c*60 + r*20 + cl  <->  channel c, row ty+r-1, col x0+cl-1
  // OOB coords are clamped here and validity-masked at consumption.
#pragma unroll
  for (int j = 0; j < 15; ++j) {
    int q   = j * 32 + lane;                  // 15*32 == 480 exactly
    int c   = q / 60;
    int rem = q % 60;
    int r   = rem / 20;
    int cl  = rem % 20;
    int gy  = iclamp(ty + r - 1, 0, HH - 1);
    int gx  = iclamp(x0 + cl - 1, 0, WW - 1);
    unsigned voff = (unsigned)((c * HW + gy * WW + gx) * 4);
    unsigned ldsa = swbase + (unsigned)(q * 4);
    asm volatile("global_load_async_to_lds_b32 %0, %1, %2"
                 :: "v"(ldsa), "v"(voff), "s"(gbase) : "memory");
  }

  // ---- load prepacked A fragments while the async copy is in flight ----
  const v16h* Wf = (const v16h*)wpk;
  const int fbase = (t * 2) * 3 * 32;
  v16h a0[3], a1f[3];
#pragma unroll
  for (int kc = 0; kc < 3; ++kc) {
    a0[kc]  = Wf[fbase + kc * 32 + lane];
    a1f[kc] = Wf[fbase + (3 + kc) * 32 + lane];
  }

  // hint feature rows into cache for the sampling phase
  __builtin_prefetch(feat_in + ((ty + 1 < HH) ? (ty + 1) * WW + x : p), 0, 1);

  // ---- wait for the tile, then build B fragments from LDS ----
  asm volatile("s_wait_asynccnt 0x0" ::: "memory");

  // B 32x16 f16: lane<16 -> K base 0, lane>=16 -> K base 16, elem i -> K+i
  U16h bfr[3];
  const bool lo = (lane < 16);
  if (lo) build_B<0>(bfr, sw, n, ty, x);
  else    build_B<16>(bfr, sw, n, ty, x);

  // ---- WMMA accumulate: C(32x16) = W(32x96) * patches(96x16) ----
  U8f c0 = {}, c1 = {};
#pragma unroll
  for (int kc = 0; kc < 3; ++kc) {
    c0.v = __builtin_amdgcn_wmma_f32_16x16x32_f16(
        false, a0[kc],  false, bfr[kc].v, (short)0, c0.v, false, false);
    c1.v = __builtin_amdgcn_wmma_f32_16x16x32_f16(
        false, a1f[kc], false, bfr[kc].v, (short)0, c1.v, false, false);
  }

  // ---- gather all 32 M-rows of this pixel into one lane via xor-16 shuffle
  // C layout: lane<16 vgpr r -> M=r ; lane>=16 vgpr r -> M=8+r (tile0), +16 (tile1)
  float lowC0[8], hiC0[8], lowC1[8], hiC1[8];
#pragma unroll
  for (int r = 0; r < 8; ++r) {
    float m0 = c0.e[r], o0 = __shfl_xor(m0, 16);
    lowC0[r] = lo ? m0 : o0;  hiC0[r] = lo ? o0 : m0;
    float m1 = c1.e[r], o1 = __shfl_xor(m1, 16);
    lowC1[r] = lo ? m1 : o1;  hiC1[r] = lo ? o1 : m1;
  }

  // ---- affinity: sigmoid then normalize over the 9 taps
  float aff[9];
  float s = 1e-8f;
#pragma unroll
  for (int k = 0; k < 9; ++k) {
    float raw = (k < 8 ? lowC0[k] : hiC0[0]) + aff_w_b[t * 9 + k];
    float sg = 1.f / (1.f + __expf(-raw));
    aff[k] = sg; s += sg;
  }
  const float inv = 1.f / s;

  // ---- offsets: conv channels 0..15 live at M rows 9..24
  float offc[16];
#pragma unroll
  for (int ch = 0; ch < 16; ++ch) {
    float raw;
    if (ch < 7)       raw = hiC0[ch + 1];     // M 9..15
    else if (ch < 15) raw = lowC1[ch - 7];    // M 16..23
    else              raw = hiC1[0];          // M 24
    offc[ch] = raw + aff_o_b[t * 16 + ch];
  }

  // ---- deformable 9-tap bilinear sampling (tap 4 = identity offset 0)
  // Branch-free: load clamped coordinate, then select 0 for OOB.
  float acc = 0.f;
#pragma unroll
  for (int k = 0; k < 9; ++k) {
    float oy = 0.f, ox = 0.f;
    if (k < 4)      { oy = offc[2 * k];     ox = offc[2 * k + 1]; }
    else if (k > 4) { oy = offc[2 * k - 2]; ox = offc[2 * k - 1]; }
    float ys = (float)ty + (float)(k / 3 - 1) + oy;
    float xs = (float)x  + (float)(k % 3 - 1) + ox;
    float y0f = floorf(ys), x0f = floorf(xs);
    int yi = (int)y0f, xi = (int)x0f;
    float wy = ys - y0f, wx = xs - x0f;

    int yc0 = iclamp(yi, 0, HH - 1),     yc1 = iclamp(yi + 1, 0, HH - 1);
    int xc0 = iclamp(xi, 0, WW - 1),     xc1 = iclamp(xi + 1, 0, WW - 1);
    bool oky0 = (yi >= 0)     & (yi < HH);
    bool oky1 = (yi + 1 >= 0) & (yi + 1 < HH);
    bool okx0 = (xi >= 0)     & (xi < WW);
    bool okx1 = (xi + 1 >= 0) & (xi + 1 < WW);

    float r00 = feat_in[yc0 * WW + xc0];
    float r01 = feat_in[yc0 * WW + xc1];
    float r10 = feat_in[yc1 * WW + xc0];
    float r11 = feat_in[yc1 * WW + xc1];
    float v00 = (oky0 & okx0) ? r00 : 0.f;
    float v01 = (oky0 & okx1) ? r01 : 0.f;
    float v10 = (oky1 & okx0) ? r10 : 0.f;
    float v11 = (oky1 & okx1) ? r11 : 0.f;

    float samp = v00 * (1.f - wy) * (1.f - wx) + v01 * (1.f - wy) * wx
               + v10 * wy * (1.f - wx)         + v11 * wy * wx;
    acc += (aff[k] * inv) * samp;
  }
  if (lo) feat_out[p] = acc;
}

// =====================================================================
// Tail stage 1: x3 -> proj(3->6) -> BN -> LeakyReLU -> concat(attn) ->
// per-channel 1x1 (lsk0, groups=22)  => a1 (22 planes)
// =====================================================================
__global__ __launch_bounds__(256) void nlspn_tail_a1(
    const float* __restrict__ f3, const float* __restrict__ f4, const float* __restrict__ f5,
    const float* __restrict__ attn,
    const float* __restrict__ proj_w,
    const float* __restrict__ bn_gamma, const float* __restrict__ bn_beta,
    const float* __restrict__ bn_mean,  const float* __restrict__ bn_var,
    const float* __restrict__ lsk0_w,   const float* __restrict__ lsk0_b,
    float* __restrict__ a1)
{
  int p = blockIdx.x * 256 + threadIdx.x;
  if (p >= HW) return;
  float x3[3] = { f3[p], f4[p], f5[p] };
  float sf[6];
#pragma unroll
  for (int j = 0; j < 6; ++j) {
    float s = proj_w[j * 3 + 0] * x3[0] + proj_w[j * 3 + 1] * x3[1] + proj_w[j * 3 + 2] * x3[2];
    s = (s - bn_mean[j]) * rsqrtf(bn_var[j] + 1e-5f) * bn_gamma[j] + bn_beta[j];
    sf[j] = (s >= 0.f) ? s : 0.2f * s;
  }
#pragma unroll
  for (int ch = 0; ch < 22; ++ch) {
    float xv = (ch < 16) ? attn[ch * HW + p] : sf[ch - 16];
    a1[ch * HW + p] = xv * lsk0_w[ch] + lsk0_b[ch];
  }
}

// =====================================================================
// Tail stage 2: depthwise 3x3 (lsks) -> two 1x1 (22->11) -> channel
// mean/max  => a1v, a2v (11 planes each), agg (2 planes)
// =====================================================================
__global__ __launch_bounds__(256) void nlspn_tail_b2(
    const float* __restrict__ a1,
    const float* __restrict__ lsks_w, const float* __restrict__ lsks_b,
    const float* __restrict__ lsk1_w, const float* __restrict__ lsk1_b,
    const float* __restrict__ lsk2_w, const float* __restrict__ lsk2_b,
    float* __restrict__ a1v, float* __restrict__ a2v, float* __restrict__ agg)
{
  int p = blockIdx.x * 256 + threadIdx.x;
  if (p >= HW) return;
  int y = p / WW, x = p % WW;
  float a1loc[22], a2loc[22];
#pragma unroll
  for (int ch = 0; ch < 22; ++ch) a1loc[ch] = a1[ch * HW + p];
#pragma unroll
  for (int ch = 0; ch < 22; ++ch) a2loc[ch] = lsks_b[ch];
#pragma unroll
  for (int pos = 0; pos < 9; ++pos) {
    int yy = y + pos / 3 - 1, xx = x + pos % 3 - 1;
    bool ok = (yy >= 0) & (yy < HH) & (xx >= 0) & (xx < WW);
    int off = iclamp(yy, 0, HH - 1) * WW + iclamp(xx, 0, WW - 1);
#pragma unroll
    for (int ch = 0; ch < 22; ++ch) {
      float w = ok ? lsks_w[ch * 9 + pos] : 0.f;
      a2loc[ch] += a1[ch * HW + off] * w;
    }
  }
  float mean = 0.f, mx = -3.4e38f;
#pragma unroll
  for (int j = 0; j < 11; ++j) {
    float s1 = lsk1_b[j], s2 = lsk2_b[j];
#pragma unroll
    for (int ch = 0; ch < 22; ++ch) {
      s1 += lsk1_w[j * 22 + ch] * a1loc[ch];
      s2 += lsk2_w[j * 22 + ch] * a2loc[ch];
    }
    a1v[j * HW + p] = s1;
    a2v[j * HW + p] = s2;
    mean += s1 + s2;
    mx = fmaxf(mx, fmaxf(s1, s2));
  }
  agg[p]      = mean * (1.f / 22.f);
  agg[HW + p] = mx;
}

// =====================================================================
// Tail stage 3: 3x3 conv (2->2) + sigmoid -> blend -> 1x1 (11->3) ->
// weighted sum with x3  => final output
// =====================================================================
__global__ __launch_bounds__(256) void nlspn_tail_b3(
    const float* __restrict__ agg,
    const float* __restrict__ lsksq_w, const float* __restrict__ lsksq_b,
    const float* __restrict__ a1v, const float* __restrict__ a2v,
    const float* __restrict__ lskc_w, const float* __restrict__ lskc_b,
    const float* __restrict__ f3, const float* __restrict__ f4, const float* __restrict__ f5,
    float* __restrict__ out)
{
  int p = blockIdx.x * 256 + threadIdx.x;
  if (p >= HW) return;
  int y = p / WW, x = p % WW;
  float s0 = lsksq_b[0], s1 = lsksq_b[1];
#pragma unroll
  for (int pos = 0; pos < 9; ++pos) {
    int yy = y + pos / 3 - 1, xx = x + pos % 3 - 1;
    bool ok = (yy >= 0) & (yy < HH) & (xx >= 0) & (xx < WW);
    int off = iclamp(yy, 0, HH - 1) * WW + iclamp(xx, 0, WW - 1);
#pragma unroll
    for (int c = 0; c < 2; ++c) {
      float v = agg[c * HW + off];
      s0 += v * (ok ? lsksq_w[(0 * 2 + c) * 9 + pos] : 0.f);
      s1 += v * (ok ? lsksq_w[(1 * 2 + c) * 9 + pos] : 0.f);
    }
  }
  float sg0 = 1.f / (1.f + __expf(-s0));
  float sg1 = 1.f / (1.f + __expf(-s1));

  float at3[3] = { lskc_b[0], lskc_b[1], lskc_b[2] };
#pragma unroll
  for (int j = 0; j < 11; ++j) {
    float a = a1v[j * HW + p] * sg0 + a2v[j * HW + p] * sg1;
    at3[0] += lskc_w[0 * 11 + j] * a;
    at3[1] += lskc_w[1 * 11 + j] * a;
    at3[2] += lskc_w[2 * 11 + j] * a;
  }
  out[p] = f3[p] * at3[0] + f4[p] * at3[1] + f5[p] * at3[2];
}

// =====================================================================
extern "C" void kernel_launch(void* const* d_in, const int* in_sizes, int n_in,
                              void* d_out, int out_size, void* d_ws, size_t ws_size,
                              hipStream_t stream) {
  const float* feat_init = (const float*)d_in[0];
  const float* guidance  = (const float*)d_in[1];
  const float* attn      = (const float*)d_in[2];
  // d_in[3] = feat_fix (unused by the reference forward)
  const float* aff_w_w   = (const float*)d_in[4];
  const float* aff_w_b   = (const float*)d_in[5];
  const float* aff_o_w   = (const float*)d_in[6];
  const float* aff_o_b   = (const float*)d_in[7];
  const float* proj_w    = (const float*)d_in[8];
  const float* bn_gamma  = (const float*)d_in[9];
  const float* bn_beta   = (const float*)d_in[10];
  const float* bn_mean   = (const float*)d_in[11];
  const float* bn_var    = (const float*)d_in[12];
  const float* lsk0_w    = (const float*)d_in[13];
  const float* lsk0_b    = (const float*)d_in[14];
  const float* lsks_w    = (const float*)d_in[15];
  const float* lsks_b    = (const float*)d_in[16];
  const float* lsk1_w    = (const float*)d_in[17];
  const float* lsk1_b    = (const float*)d_in[18];
  const float* lsk2_w    = (const float*)d_in[19];
  const float* lsk2_b    = (const float*)d_in[20];
  const float* lsksq_w   = (const float*)d_in[21];
  const float* lsksq_b   = (const float*)d_in[22];
  const float* lskc_w    = (const float*)d_in[23];
  const float* lskc_b    = (const float*)d_in[24];

  // ---- workspace layout (floats) ----
  float* ws = (float*)d_ws;
  _Float16* wpk = (_Float16*)ws;           // 18432 halfs (36 KB), keep 10240 floats reserved
  float* base = ws + 10240;
  float* step[TSTEPS];
  for (int t = 0; t < TSTEPS; ++t) step[t] = base + (size_t)t * HW;
  float* a1  = base + (size_t)TSTEPS * HW;        // 22 planes
  float* a1v = a1  + (size_t)22 * HW;             // 11 planes
  float* a2v = a1v + (size_t)11 * HW;             // 11 planes
  float* agg = a2v + (size_t)11 * HW;             // 2 planes

  // 1) pack WMMA weight fragments
  nlspn_pack_weights<<<72, 256, 0, stream>>>(aff_w_w, aff_o_w, wpk);

  // 2) six fused conv(WMMA) + DCN propagation steps (sequential scan)
  const int nwaves = HW / 16;                  // 26752
  const int ablocks = nwaves / 8;              // 3344 (blockDim 256 = 8 waves)
  const float* fin = feat_init;
  for (int t = 0; t < TSTEPS; ++t) {
    nlspn_step_kernel<<<ablocks, 256, 0, stream>>>(
        guidance, fin, wpk, aff_w_b, aff_o_b, step[t], t);
    fin = step[t];
  }

  // 3) tail
  const int pblocks = HW / 256;                // 1672
  nlspn_tail_a1<<<pblocks, 256, 0, stream>>>(
      step[3], step[4], step[5], attn, proj_w,
      bn_gamma, bn_beta, bn_mean, bn_var, lsk0_w, lsk0_b, a1);
  nlspn_tail_b2<<<pblocks, 256, 0, stream>>>(
      a1, lsks_w, lsks_b, lsk1_w, lsk1_b, lsk2_w, lsk2_b, a1v, a2v, agg);
  nlspn_tail_b3<<<pblocks, 256, 0, stream>>>(
      agg, lsksq_w, lsksq_b, a1v, a2v, lskc_w, lskc_b,
      step[3], step[4], step[5], (float*)d_out);
}
